// VLAModel_10419590660411
// MI455X (gfx1250) — matrix operand order repaired and checked by reading, not compiled, not run on hardware
//
#include <hip/hip_runtime.h>
#include <hip/hip_bf16.h>
#include <math.h>
#include <stdint.h>

// ---------------- problem constants ----------------
#define LAYERS   6
#define BATCH    8
#define TLEN     512
#define DMODEL   2048
#define DSTATE   128
#define DCONV    4
#define DINNER   4096              // EXPAND * DMODEL
#define MTOK     (BATCH * TLEN)    // 4096 tokens

// ---------------- GEMM tiling ----------------
#define BM   128
#define BN   128
#define BK   64                    // K per TDM tile (bf16 elements)
#define LDSROWB 144                // LDS bytes per tile row: BK*2 + 16 pad (TDM pad)
#define TILEB   (128 * LDSROWB)    // one 128-row tile in LDS = 18432 B

typedef __attribute__((ext_vector_type(16))) __bf16 v16bf;
typedef __attribute__((ext_vector_type(8)))  float  v8f;
typedef __attribute__((ext_vector_type(4)))  unsigned int u32x4;
typedef __attribute__((ext_vector_type(8)))  int i32x8;
typedef __attribute__((ext_vector_type(4)))  int i32x4;

__device__ __forceinline__ int rfl(unsigned v) {
    return __builtin_amdgcn_readfirstlane((int)v);
}

// =====================================================================
// TDM: DMA a 32-row x BK-col bf16 tile (row stride K elems) into LDS.
// D# per CDNA5 ISA ch.8: group0 = {count|flags, lds_addr, global_addr lo,
// global_addr hi | type=2}; group1 carries dims/strides/pad. Pad: 16B after
// every 128B row -> 144B LDS row stride (conflict-free fragment reads).
// This toolchain exposes the 6-arg builtin (extra i32x8 group, zero-filled).
// =====================================================================
__device__ __forceinline__ void tdm_issue(const __bf16* gsrc, int row0, int K,
                                          int k0, unsigned lds_off) {
    uint64_t ga = (uint64_t)(uintptr_t)(gsrc + (uint64_t)row0 * K + k0);
    u32x4 g0;
    g0[0] = (unsigned)rfl(1u);                                  // count=1 valid D#
    g0[1] = (unsigned)rfl(lds_off);                             // LDS byte address
    g0[2] = (unsigned)rfl((unsigned)ga);                        // global addr [31:0]
    g0[3] = (unsigned)rfl(((unsigned)(ga >> 32) & 0x01FFFFFFu)  // global addr [56:32]
                          | 0x80000000u);                       // type=2 ("image")
    i32x8 g1;
    g1[0] = rfl((1u << 16)      // data_size = 2 bytes
              | (1u << 20)      // pad_enable
              | (4u << 22)      // pad_interval: 32 DWORDs (=one 128B row)
              | (3u << 25));    // pad_amount: 4 DWORDs (16B)
    g1[1] = rfl(((unsigned)K & 0xFFFFu) << 16);                 // tensor_dim0 lo16
    g1[2] = rfl(((unsigned)K >> 16) | (32u << 16));             // tensor_dim0 hi | tensor_dim1=32
    g1[3] = rfl((unsigned)BK << 16);                            // tile_dim0 = BK
    g1[4] = rfl(32u);                                           // tile_dim1 = 32 rows
    g1[5] = rfl((unsigned)K);                                   // tensor_dim0_stride lo
    g1[6] = 0;
    g1[7] = 0;
    i32x4 gz  = {0, 0, 0, 0};
    i32x8 gz8 = {0, 0, 0, 0, 0, 0, 0, 0};
    __builtin_amdgcn_tensor_load_to_lds(g0, g1, gz, gz, gz8, 0);
}

// =====================================================================
// LayerNorm: one block per token, 256 threads -> bf16 output
// =====================================================================
__global__ __launch_bounds__(256)
void ln_kernel(const float* __restrict__ x, const float* __restrict__ g,
               const float* __restrict__ b, __bf16* __restrict__ out) {
    const int row = blockIdx.x;
    const int tid = threadIdx.x;
    const float* xr = x + (size_t)row * DMODEL;

    float s = 0.f, s2 = 0.f;
    #pragma unroll
    for (int j = tid; j < DMODEL; j += 256) {
        float v = xr[j];
        s += v; s2 += v * v;
    }
    __shared__ float red0[8], red1[8];
    #pragma unroll
    for (int off = 16; off > 0; off >>= 1) {
        s  += __shfl_down(s,  off, 32);
        s2 += __shfl_down(s2, off, 32);
    }
    if ((tid & 31) == 0) { red0[tid >> 5] = s; red1[tid >> 5] = s2; }
    __syncthreads();
    __shared__ float s_mu, s_rstd;
    if (tid == 0) {
        float S = 0.f, S2 = 0.f;
        #pragma unroll
        for (int i = 0; i < 8; i++) { S += red0[i]; S2 += red1[i]; }
        float mu  = S * (1.0f / DMODEL);
        float var = S2 * (1.0f / DMODEL) - mu * mu;
        s_mu = mu; s_rstd = rsqrtf(var + 1e-5f);
    }
    __syncthreads();
    const float mu = s_mu, rstd = s_rstd;
    __bf16* orow = out + (size_t)row * DMODEL;
    #pragma unroll
    for (int j = tid; j < DMODEL; j += 256)
        orow[j] = (__bf16)((xr[j] - mu) * rstd * g[j] + b[j]);
}

// =====================================================================
// bf16 WMMA GEMM with TDM double-buffered tile pipeline.
//   C[M,N] = A[M,K] * W[N,K]^T
//   MODE 0: C = acc (fp32)
//   MODE 1: softplus(acc + bias[n]) row-reduced via atomicAdd into redout[m]
//   MODE 2: C = acc + resid[m,n]
// 256 threads / 8 waves. Waves 0-3 DMA 32-row slices of the A tile,
// waves 4-7 the W tile. Wave tile 32x64 => 2x4 fragments, 16 WMMA / K-iter.
// =====================================================================
template <int MODE>
__global__ __launch_bounds__(256)
void gemm_bf16_wmma(const __bf16* __restrict__ A, const __bf16* __restrict__ W,
                    float* __restrict__ C, const float* __restrict__ bias,
                    const float* __restrict__ resid, float* __restrict__ redout,
                    int M, int N, int K) {
    __shared__ __align__(16) unsigned char smem[2][2][TILEB]; // [buf][A|B][tile]

    const int tid  = threadIdx.x;
    const int lane = tid & 31;
    const int wid  = tid >> 5;
    const int wm   = (wid & 3) * 32;   // wave M offset within block
    const int wn   = (wid >> 2) * 64;  // wave N offset within block
    const int bm0  = blockIdx.y * BM;
    const int bn0  = blockIdx.x * BN;
    const int hsel = lane >> 4;
    const int l16  = lane & 15;

    // this wave's DMA job: tile select (A vs W) + 32-row slice
    const int ab      = wid >> 2;            // 0: A tile, 1: W tile
    const int wslice  = wid & 3;
    const __bf16* src = ab ? W : A;
    const int row0    = (ab ? bn0 : bm0) + wslice * 32;
    const unsigned lds_base = (unsigned)(uintptr_t)&smem[0][0][0];

    v8f acc[2][4];
    #pragma unroll
    for (int i = 0; i < 2; i++)
        #pragma unroll
        for (int j = 0; j < 4; j++)
            acc[i][j] = v8f{0.f,0.f,0.f,0.f,0.f,0.f,0.f,0.f};

    const int ntiles = K / BK;
    // prologue: DMA tile 0 into buffer 0
    tdm_issue(src, row0, K, 0,
              lds_base + (unsigned)(ab * TILEB) + (unsigned)(wslice * 32 * LDSROWB));

    for (int it = 0; it < ntiles; ++it) {
        const int buf = it & 1;
        __builtin_amdgcn_s_wait_tensorcnt(0);   // this wave's DMA done
        __syncthreads();                        // everyone's DMA done, prev compute done
        if (it + 1 < ntiles)                    // DMA next tile into other buffer
            tdm_issue(src, row0, K, (it + 1) * BK,
                      lds_base + (unsigned)(((buf ^ 1) * 2 + ab) * TILEB)
                               + (unsigned)(wslice * 32 * LDSROWB));

        const unsigned char* atile = &smem[buf][0][0];
        const unsigned char* btile = &smem[buf][1][0];
        #pragma unroll
        for (int kk = 0; kk < BK; kk += 32) {
            v16bf afr[2], bfr[4];
            #pragma unroll
            for (int fm = 0; fm < 2; fm++) {
                const __bf16* p = (const __bf16*)(atile + (wm + fm * 16 + l16) * LDSROWB)
                                  + kk + hsel * 8;
                #pragma unroll
                for (int i = 0; i < 8; i++) { afr[fm][i] = p[i]; afr[fm][i + 8] = p[i + 16]; }
            }
            #pragma unroll
            for (int fn = 0; fn < 4; fn++) {
                const __bf16* p = (const __bf16*)(btile + (wn + fn * 16 + l16) * LDSROWB)
                                  + kk + hsel * 8;
                #pragma unroll
                for (int i = 0; i < 8; i++) { bfr[fn][i] = p[i]; bfr[fn][i + 8] = p[i + 16]; }
            }
            #pragma unroll
            for (int fm = 0; fm < 2; fm++)
                #pragma unroll
                for (int fn = 0; fn < 4; fn++)
                    acc[fm][fn] = __builtin_amdgcn_wmma_f32_16x16x32_bf16(
                        false, afr[fm], false, bfr[fn],
                        (short)0, acc[fm][fn], false, false);
        }
        __syncthreads();   // all waves done reading buf before it is re-filled
    }

    // ---- epilogue: D layout: VGPR r -> m = r + hsel*8, n = l16 ----
    #pragma unroll
    for (int fm = 0; fm < 2; fm++) {
        #pragma unroll
        for (int r = 0; r < 8; r++) {
            const int m = bm0 + wm + fm * 16 + hsel * 8 + r;
            if (MODE == 1) {
                float srow = 0.f;
                #pragma unroll
                for (int fn = 0; fn < 4; fn++) {
                    const int n = bn0 + wn + fn * 16 + l16;
                    float v = acc[fm][fn][r] + bias[n];
                    srow += (v > 20.f) ? v : log1pf(__expf(v)); // softplus
                }
                atomicAdd(&redout[m], srow);
            } else {
                #pragma unroll
                for (int fn = 0; fn < 4; fn++) {
                    const int n = bn0 + wn + fn * 16 + l16;
                    float v = acc[fm][fn][r];
                    if (MODE == 2) v += resid[(size_t)m * N + n];
                    C[(size_t)m * N + n] = v;
                }
            }
        }
    }
}

// =====================================================================
// Causal depthwise conv1d (k=4) + bias + SiLU -> bf16 xc
// =====================================================================
__global__ __launch_bounds__(256)
void conv_silu_kernel(const float* __restrict__ xar, const float* __restrict__ w,
                      const float* __restrict__ cb, __bf16* __restrict__ xc) {
    const size_t idx = (size_t)blockIdx.x * 256 + threadIdx.x; // bt * DINNER + e
    const int    e   = (int)(idx & (DINNER - 1));
    const size_t bt  = idx >> 12;
    const int    t   = (int)(bt & (TLEN - 1));
    float acc = cb[e];
    #pragma unroll
    for (int k = 0; k < DCONV; k++) {
        const int tt = t - (DCONV - 1) + k;
        if (tt >= 0)
            acc += w[e * DCONV + k] * xar[(bt - (size_t)(t - tt)) * (2 * DINNER) + e];
    }
    xc[idx] = (__bf16)(acc / (1.f + __expf(-acc)));   // SiLU
}

// =====================================================================
// Selective-scan: 1 block per batch, 128 threads = one state each (fp32)
// =====================================================================
__global__ __launch_bounds__(128)
void scan_kernel(const float* __restrict__ bc, const float* __restrict__ dsum,
                 const float* __restrict__ A_log, float* __restrict__ ysc) {
    const int b = blockIdx.x;
    const int s = threadIdx.x;
    const float As = -__expf(A_log[s]);
    float h = 0.f;
    __shared__ float wpart[4];
    for (int t = 0; t < TLEN; t++) {
        const size_t bt = (size_t)b * TLEN + t;
        const float dm = dsum[bt] * (1.0f / DINNER);
        const float a = __expf(dm * As);
        const float* row = bc + bt * (2 * DSTATE);
        h = h * a + row[s];
        float p = h * row[DSTATE + s];
        #pragma unroll
        for (int off = 16; off > 0; off >>= 1) p += __shfl_down(p, off, 32);
        if ((s & 31) == 0) wpart[s >> 5] = p;
        __syncthreads();
        if (s == 0) ysc[bt] = wpart[0] + wpart[1] + wpart[2] + wpart[3];
        __syncthreads();
    }
}

// =====================================================================
// y = (y_scan + D_skip * xc) * silu(res)  -> bf16 (feeds W_out GEMM)
// =====================================================================
__global__ __launch_bounds__(256)
void combine_kernel(const float* __restrict__ ysc, const __bf16* __restrict__ xc,
                    const float* __restrict__ xar, const float* __restrict__ Dsk,
                    __bf16* __restrict__ ycmb) {
    const size_t idx = (size_t)blockIdx.x * 256 + threadIdx.x;
    const int    e   = (int)(idx & (DINNER - 1));
    const size_t bt  = idx >> 12;
    const float res  = xar[bt * (2 * DINNER) + DINNER + e];
    const float sres = res / (1.f + __expf(-res));
    ycmb[idx] = (__bf16)((ysc[bt] + Dsk[e] * (float)xc[idx]) * sres);
}

// =====================================================================
// fp32 -> bf16 weight conversion (per-layer, once)
// =====================================================================
__global__ __launch_bounds__(256)
void cvt_kernel(const float* __restrict__ s, __bf16* __restrict__ d, size_t n) {
    const size_t i = ((size_t)blockIdx.x * 256 + threadIdx.x) * 4;
    if (i + 3 < n) {
        float4 v = *(const float4*)(s + i);
        d[i]     = (__bf16)v.x; d[i + 1] = (__bf16)v.y;
        d[i + 2] = (__bf16)v.z; d[i + 3] = (__bf16)v.w;
    }
}

__global__ void zero_kernel(float* p, int n) {
    int i = blockIdx.x * 256 + threadIdx.x;
    if (i < n) p[i] = 0.f;
}

// =====================================================================
// Host orchestration
// =====================================================================
extern "C" void kernel_launch(void* const* d_in, const int* in_sizes, int n_in,
                              void* d_out, int out_size, void* d_ws, size_t ws_size,
                              hipStream_t stream) {
    const float* x      = (const float*)d_in[0];
    const float* ln_g   = (const float*)d_in[1];
    const float* ln_b   = (const float*)d_in[2];
    const float* W_in   = (const float*)d_in[3];
    const float* conv_w = (const float*)d_in[4];
    const float* conv_b = (const float*)d_in[5];
    const float* W_xp   = (const float*)d_in[6];
    const float* W_dt   = (const float*)d_in[7];
    const float* b_dt   = (const float*)d_in[8];
    const float* A_log  = (const float*)d_in[9];
    const float* D_skip = (const float*)d_in[10];
    const float* W_out  = (const float*)d_in[11];
    float* out = (float*)d_out;

    unsigned char* ws = (unsigned char*)d_ws;
    size_t off = 0;
    auto allocB = [&](size_t bytes) {
        void* p = ws + off;
        off = (off + bytes + 255) & ~(size_t)255;
        return p;
    };
    const size_t NW_IN = (size_t)2 * DINNER * DMODEL;   // 16.78M
    const size_t NW_XP = (size_t)2 * DSTATE * DINNER;   //  1.05M
    const size_t NW_DT = (size_t)DINNER * DINNER;       // 16.78M
    const size_t NW_OUT= (size_t)DMODEL * DINNER;       //  8.39M

    __bf16* x_n  = (__bf16*)allocB((size_t)MTOK * DMODEL * 2);
    float*  xar  = (float*) allocB((size_t)MTOK * 2 * DINNER * 4);
    __bf16* xc   = (__bf16*)allocB((size_t)MTOK * DINNER * 2);
    float*  bcs  = (float*) allocB((size_t)MTOK * 2 * DSTATE * 4);
    float*  dsum = (float*) allocB((size_t)MTOK * 4);
    float*  ysc  = (float*) allocB((size_t)MTOK * 4);
    __bf16* ycmb = (__bf16*)allocB((size_t)MTOK * DINNER * 2);
    float*  hA   = (float*) allocB((size_t)MTOK * DMODEL * 4);
    float*  hB   = (float*) allocB((size_t)MTOK * DMODEL * 4);
    __bf16* wb_in  = (__bf16*)allocB(NW_IN  * 2);
    __bf16* wb_xp  = (__bf16*)allocB(NW_XP  * 2);
    __bf16* wb_dt  = (__bf16*)allocB(NW_DT  * 2);
    __bf16* wb_out = (__bf16*)allocB(NW_OUT * 2);

    const dim3 blk256(256), blk128(128);
    const float* hin = x;
    for (int l = 0; l < LAYERS; l++) {
        float* hout = (l == LAYERS - 1) ? out : ((l & 1) ? hB : hA);

        // per-layer weight slices -> bf16
        cvt_kernel<<<(unsigned)(NW_IN  / 1024), blk256, 0, stream>>>(W_in  + (size_t)l * NW_IN,  wb_in,  NW_IN);
        cvt_kernel<<<(unsigned)(NW_XP  / 1024), blk256, 0, stream>>>(W_xp  + (size_t)l * NW_XP,  wb_xp,  NW_XP);
        cvt_kernel<<<(unsigned)(NW_DT  / 1024), blk256, 0, stream>>>(W_dt  + (size_t)l * NW_DT,  wb_dt,  NW_DT);
        cvt_kernel<<<(unsigned)(NW_OUT / 1024), blk256, 0, stream>>>(W_out + (size_t)l * NW_OUT, wb_out, NW_OUT);

        ln_kernel<<<MTOK, blk256, 0, stream>>>(hin, ln_g + l * DMODEL,
                                               ln_b + l * DMODEL, x_n);

        gemm_bf16_wmma<0><<<dim3(2 * DINNER / BN, MTOK / BM), blk256, 0, stream>>>(
            x_n, wb_in, xar, nullptr, nullptr, nullptr, MTOK, 2 * DINNER, DMODEL);

        conv_silu_kernel<<<(unsigned)((MTOK * (size_t)DINNER) / 256), blk256, 0, stream>>>(
            xar, conv_w + (size_t)l * DINNER * DCONV, conv_b + l * DINNER, xc);

        gemm_bf16_wmma<0><<<dim3(2 * DSTATE / BN, MTOK / BM), blk256, 0, stream>>>(
            xc, wb_xp, bcs, nullptr, nullptr, nullptr, MTOK, 2 * DSTATE, DINNER);

        zero_kernel<<<MTOK / 256, blk256, 0, stream>>>(dsum, MTOK);

        gemm_bf16_wmma<1><<<dim3(DINNER / BN, MTOK / BM), blk256, 0, stream>>>(
            xc, wb_dt, nullptr, b_dt + l * DINNER, nullptr, dsum, MTOK, DINNER, DINNER);

        scan_kernel<<<BATCH, blk128, 0, stream>>>(bcs, dsum, A_log + l * DSTATE, ysc);

        combine_kernel<<<(unsigned)((MTOK * (size_t)DINNER) / 256), blk256, 0, stream>>>(
            ysc, xc, xar, D_skip + l * DINNER, ycmb);

        gemm_bf16_wmma<2><<<dim3(DMODEL / BN, MTOK / BM), blk256, 0, stream>>>(
            ycmb, wb_out, hout, nullptr, hin, nullptr, MTOK, DMODEL, DINNER);

        hin = hout;
    }
}
